// VectorQuantizer_575525617951
// MI455X (gfx1250) — matrix operand order repaired and checked by reading, not compiled, not run on hardware
//
#include <hip/hip_runtime.h>
#include <hip/hip_bf16.h>

typedef __attribute__((ext_vector_type(16))) _Float16 v16h;
typedef __attribute__((ext_vector_type(8)))  _Float16 v8h;
typedef __attribute__((ext_vector_type(8)))  float    v8f;
typedef __attribute__((ext_vector_type(4)))  unsigned u32x4;
typedef __attribute__((ext_vector_type(8)))  unsigned u32x8;

#define NUM_EMB   1024
#define EMB_DIM   256
#define HWPIX     1024                    // H*W
#define SLAB      (EMB_DIM * HWPIX)       // 262144 elems per image
#define OUT_HALF  (32 * SLAB)             // 8388608 elems per output tensor

#define CHUNK     128                     // codes staged per TDM transfer
#define NCHUNK    (NUM_EMB / CHUNK)       // 8
#define ROW_BYTES (EMB_DIM * 2)           // 512 B per f16 code row
#define PAD_BYTES 16                      // TDM pad: 4 dwords per row
#define LDS_ROW   (ROW_BYTES + PAD_BYTES) // 528 B -> bank-conflict-free B reads
#define EBUF_SZ   (CHUNK * LDS_ROW)       // 67584 B per buffer

// ---------------------------------------------------------------------------
// Prep: codebook f32 -> f16 (row-major [N][C] matches WMMA B layout) + ||e||^2.
// ---------------------------------------------------------------------------
__global__ __launch_bounds__(256) void vq_prep(const float* __restrict__ emb,
                                               _Float16* __restrict__ e16,
                                               float* __restrict__ enorm) {
    const int n = blockIdx.x, t = threadIdx.x;
    float v = emb[n * EMB_DIM + t];
    e16[n * EMB_DIM + t] = (_Float16)v;

    __shared__ float red[256];
    red[t] = v * v;
    __syncthreads();
    for (int s = 128; s > 0; s >>= 1) {
        if (t < s) red[t] += red[t + s];
        __syncthreads();
    }
    if (t == 0) enorm[n] = red[0];
}

// ---------------------------------------------------------------------------
// TDM: issue tensor_load_to_lds for one 128-code chunk (64 KB) with per-row
// 16B LDS padding. Descriptor per CDNA5 D# spec (groups 0 and 1, 1-D tile).
// ---------------------------------------------------------------------------
static __device__ __forceinline__ void tdm_load_chunk(const _Float16* gsrc,
                                                      unsigned lds_addr) {
    const unsigned long long ga = (unsigned long long)(uintptr_t)gsrc;
    u32x4 g0;
    g0.x = 1u;                                   // count=1, user descriptor
    g0.y = lds_addr;                             // LDS dest byte address
    g0.z = (unsigned)ga;                         // global_addr[31:0]
    g0.w = (unsigned)(ga >> 32) | 0x80000000u;   // global_addr[56:32] | type=2
    u32x8 g1;
    g1[0] = (1u << 16)     // data_size = 2 bytes
          | (1u << 20)     // pad_enable
          | (6u << 22)     // pad_interval: every 128 dwords (= one 512B row)
          | (3u << 25);    // pad_amount: 4 dwords (16 B)
    g1[1] = 0u;                                  // tensor_dim0[15:0]=0 (0x40000)
    g1[2] = 0x00010004u;                         // tensor_dim0[31:16]=4, tensor_dim1=1
    g1[3] = 0x80000000u;                         // tile_dim0 = 32768 elems (64 KB)
    g1[4] = 0u;                                  // tile_dim1/2 unused
    g1[5] = 0x00040000u;                         // tensor_dim0_stride = 262144
    g1[6] = 0u;
    g1[7] = 0u;
    asm volatile("tensor_load_to_lds %0, %1" :: "s"(g0), "s"(g1) : "memory");
}

// ---------------------------------------------------------------------------
// Main: distance GEMM (v_wmma_f32_16x16x32_f16) with TDM-staged codebook in
// double-buffered LDS, fused argmin, then gather + dual write.
// Grid: 256 blocks x 256 threads (8 waves); wave w owns 16 pixel rows.
// ---------------------------------------------------------------------------
__global__ __launch_bounds__(256) void vq_main(const float* __restrict__ inp,
                                               const float* __restrict__ emb,
                                               const _Float16* __restrict__ e16,
                                               const float* __restrict__ enorm,
                                               float* __restrict__ out) {
    __shared__ __align__(64) char ebuf[2][EBUF_SZ];
    __shared__ int idx_s[128];

    const int t    = threadIdx.x;
    const int wave = t >> 5;
    const int lane = t & 31;
    const int half = lane >> 4;   // A/B layout: K-octet selector
    const int ln   = lane & 15;

    // Kick off chunk 0 DMA before doing anything else (overlaps A loads).
    if (wave == 0)
        tdm_load_chunk(e16, (unsigned)(uintptr_t)&ebuf[0][0]);

    // ---- A operands: lane holds matrix row M = ln (16-lane coalesced) ----
    const int mrow = blockIdx.x * 128 + wave * 16 + ln;
    const float* xrow = inp + (size_t)(mrow >> 10) * SLAB + (mrow & 1023);

    v16h A[8];
#pragma unroll
    for (int k = 0; k < 8; ++k) {
        const int c0 = k * 32 + half * 8;        // ISA 16-bit A layout, 16x32
#pragma unroll
        for (int j = 0; j < 8; ++j) {
            A[k][j]     = (_Float16)xrow[(size_t)(c0 + j)      * HWPIX];
            A[k][j + 8] = (_Float16)xrow[(size_t)(c0 + 16 + j) * HWPIX];
        }
    }

    float best_d[8];
    int   best_n[8];
#pragma unroll
    for (int v = 0; v < 8; ++v) { best_d[v] = 3.0e38f; best_n[v] = 0; }

    if (wave == 0) __builtin_amdgcn_s_wait_tensorcnt(0);
    __syncthreads();                              // chunk 0 resident

    // ---- N loop: 8 chunks x 8 n-tiles x 8 WMMA K-steps -------------------
    for (int c = 0; c < NCHUNK; ++c) {
        // Prefetch next chunk into the other buffer (its readers all passed
        // the barrier at the end of iteration c-1).
        if (wave == 0 && c + 1 < NCHUNK)
            tdm_load_chunk(e16 + (size_t)(c + 1) * CHUNK * EMB_DIM,
                           (unsigned)(uintptr_t)&ebuf[(c + 1) & 1][0]);

        const char* bbase = &ebuf[c & 1][0];
        for (int nt = 0; nt < CHUNK / 16; ++nt) {
            const int nb = c * CHUNK + nt * 16;
            // B layout: lane = column n, channels 8*half..+7 and 16+8*half..+7.
            const char* brow = bbase + (size_t)(nt * 16 + ln) * LDS_ROW
                                     + (size_t)half * 16;
            v8f acc = {};
#pragma unroll
            for (int k = 0; k < 8; ++k) {
                const v8h* bp = (const v8h*)(brow + k * 64);  // 16B aligned
                v8h lo = bp[0], hi = bp[2];
                v16h Bv = __builtin_shufflevector(lo, hi,
                    0, 1, 2, 3, 4, 5, 6, 7, 8, 9, 10, 11, 12, 13, 14, 15);
                acc = __builtin_amdgcn_wmma_f32_16x16x32_f16(
                    false, A[k], false, Bv, (short)0, acc, false, false);
            }
            const float en = enorm[nb + ln];
#pragma unroll
            for (int v = 0; v < 8; ++v) {
                float d = en - 2.0f * acc[v];     // ||e||^2 - 2 x.e
                if (d < best_d[v]) { best_d[v] = d; best_n[v] = nb + ln; }
            }
        }
        if (wave == 0) __builtin_amdgcn_s_wait_tensorcnt(0);
        __syncthreads();                          // next chunk ready, this one free
    }

    // ---- Cross-lane argmin: row M = v + 8*half lives on 16 lanes ---------
#pragma unroll
    for (int v = 0; v < 8; ++v) {
        float d = best_d[v];
        int   n = best_n[v];
#pragma unroll
        for (int m = 1; m < 16; m <<= 1) {
            float od = __shfl_xor(d, m, 32);
            int   on = __shfl_xor(n, m, 32);
            if (od < d || (od == d && on < n)) { d = od; n = on; }
        }
        if (ln == 0) idx_s[wave * 16 + v + 8 * half] = n;
    }
    __syncthreads();

    // ---- Gather + write straight_through and quantized -------------------
    const int cg = t >> 7, r = t & 127;
    const int m  = blockIdx.x * 128 + r;
    const size_t base = (size_t)(m >> 10) * SLAB + (m & 1023);
    const float* erow = emb + (size_t)idx_s[r] * EMB_DIM;
    float* out_st = out;
    float* out_q  = out + OUT_HALF;
    for (int c = cg * 128; c < cg * 128 + 128; ++c) {
        const size_t o = base + (size_t)c * HWPIX; // fixed c: lanes contiguous
        float q  = erow[c];
        float iv = inp[o];
        out_st[o] = iv + (q - iv);
        out_q[o]  = q;
    }
}

extern "C" void kernel_launch(void* const* d_in, const int* in_sizes, int n_in,
                              void* d_out, int out_size, void* d_ws, size_t ws_size,
                              hipStream_t stream) {
    const float* inp = (const float*)d_in[0];   // [32,256,32,32] f32
    const float* emb = (const float*)d_in[1];   // [1024,256]     f32

    _Float16* e16  = (_Float16*)d_ws;                                   // 512 KB
    float*    enrm = (float*)((char*)d_ws + NUM_EMB * EMB_DIM * 2);     // 4 KB

    vq_prep<<<NUM_EMB, 256, 0, stream>>>(emb, e16, enrm);
    vq_main<<<256, 256, 0, stream>>>(inp, emb, e16, enrm, (float*)d_out);
}
